// RecurrentClassifierSJ_4269197492583
// MI455X (gfx1250) — matrix operand order repaired and checked by reading, not compiled, not run on hardware
//
#include <hip/hip_runtime.h>

typedef __attribute__((ext_vector_type(16))) __bf16 v16bf;
typedef __attribute__((ext_vector_type(8)))  float  v8f;

union FragAB { uint4 q[2]; v16bf v; };

__device__ __forceinline__ unsigned short f2bf(float f) {
  unsigned u = __float_as_uint(f);
  unsigned r = (u + 0x7FFFu + ((u >> 16) & 1u)) >> 16;  // round-to-nearest-even
  return (unsigned short)r;
}

// One 16-byte async copy global -> LDS per lane (tracked by ASYNCcnt).
// The LDS address operand is derived from the real shared-memory pointer via
// ptrtoint so the shared array escapes; the "memory" clobber then correctly
// orders these writes against the ds_load fragment reads. Low 32 bits of a
// flat LDS pointer are the wave-relative LDS byte offset (ISA 10.2).
__device__ __forceinline__ void async_b128(const void* lds_ptr, const void* g) {
  unsigned l = (unsigned)(unsigned long long)lds_ptr;
  asm volatile("global_load_async_to_lds_b128 %0, %1, off"
               :: "v"(l), "v"((unsigned long long)g)
               : "memory");
}

__device__ __forceinline__ void wait_async0() {
  asm volatile("s_wait_asynccnt 0x0" ::: "memory");
}

// ---------------------------------------------------------------------------
// Kernel 1: fp32 -> bf16 elementwise convert (x -> A[4096 x 16384])
// ---------------------------------------------------------------------------
__global__ __launch_bounds__(256)
void cvt_f32_bf16_vec4(const float* __restrict__ in,
                       unsigned short* __restrict__ out, long long n) {
  long long i = ((long long)blockIdx.x * 256 + threadIdx.x) * 4;
  if (i + 3 < n) {
    float4 f = *(const float4*)(in + i);
    uint2 p;
    p.x = (unsigned)f2bf(f.x) | ((unsigned)f2bf(f.y) << 16);
    p.y = (unsigned)f2bf(f.z) | ((unsigned)f2bf(f.w) << 16);
    *(uint2*)(out + i) = p;
  }
}

// ---------------------------------------------------------------------------
// Kernel 2: W1[F=16384][H=4096] fp32 -> W1T[H][F] bf16 (tiled transpose)
// ---------------------------------------------------------------------------
__global__ __launch_bounds__(256)
void transpose_cvt(const float* __restrict__ W1, unsigned short* __restrict__ W1T) {
  __shared__ float tile[32][33];
  const int tx = threadIdx.x, ty = threadIdx.y;
  const int x  = blockIdx.x * 32 + tx;   // H column
  const int y0 = blockIdx.y * 32;        // F row base
#pragma unroll
  for (int j = 0; j < 32; j += 8)
    tile[ty + j][tx] = W1[(size_t)(y0 + ty + j) * 4096 + x];
  __syncthreads();
  const int xo = y0 + tx;                // F col of output
  const int yo = blockIdx.x * 32 + ty;   // H row of output
#pragma unroll
  for (int j = 0; j < 32; j += 8)
    W1T[(size_t)(yo + j) * 16384 + xo] = f2bf(tile[tx][ty + j]);
}

// ---------------------------------------------------------------------------
// Kernel 3: cur1[4096][4096] = A[4096][16384](bf16) @ W1 + b1  (WMMA bf16)
// Block: 128x128 tile, 256 threads = 8 waves (4 along M x 2 along N).
// Wave: 32x64 sub-tile = 2x4 WMMA accumulators. K-step 32, double-buffered
// LDS staged with global_load_async_to_lds_b128.
// ---------------------------------------------------------------------------
#define ROWSTRIDE 80          // bytes per staged row (32 bf16 + 16B pad): bank-conflict-free
#define HALF_TILE 10240u      // 128 rows * 80 B
#define TILE_BYTES 20480u     // sA + sB

__global__ __launch_bounds__(256)
void snn_gemm_bf16(const unsigned short* __restrict__ Abf,
                   const unsigned short* __restrict__ BTbf,
                   const float* __restrict__ bias,
                   float* __restrict__ Cout) {
  __shared__ uint4 smem_q[(2 * TILE_BYTES) / 16];
  char* sm = (char*)smem_q;   // byte view for async destinations

  const int tid  = threadIdx.x;
  const int lane = tid & 31;
  const int wave = tid >> 5;
  const int wm   = wave & 3;   // 4 waves along M (32 rows each)
  const int wn   = wave >> 2;  // 2 waves along N (64 cols each)

  const int m0 = blockIdx.y * 128;
  const int n0 = blockIdx.x * 128;

  // Staging assignment: 2 threads per 32-element row (32B halves).
  const int srow  = tid >> 1;
  const int shalf = tid & 1;
  const unsigned short* gA = Abf  + (size_t)(m0 + srow) * 16384 + shalf * 16;
  const unsigned short* gB = BTbf + (size_t)(n0 + srow) * 16384 + shalf * 16;
  const unsigned lA = (unsigned)(srow * ROWSTRIDE + shalf * 32);
  const unsigned lB = HALF_TILE + lA;

  v8f acc[2][4] = {};

  // Prologue: stage k-block 0 into buffer 0.
  async_b128(sm + lA,       gA);  async_b128(sm + lA + 16u, gA + 8);
  async_b128(sm + lB,       gB);  async_b128(sm + lB + 16u, gB + 8);

  // Fragment lane offsets (ISA 7.12.2 layouts):
  // A (16x32 bf16): lanes<16 -> K{0..7,16..23}; lanes>=16 -> K{8..15,24..31}
  // B (32x16 bf16): lanes<16 -> K{0..7,16..23}; lanes>=16 -> K{8..15,24..31}
  const unsigned aoff = (lane & 16) ? 16u : 0u;
  const unsigned boff = (lane & 16) ? 16u : 0u;

  for (int kk = 0; kk < 512; ++kk) {
    const unsigned bufo = (unsigned)(kk & 1) * TILE_BYTES;
    wait_async0();        // our wave's async fills of current buffer done
    __syncthreads();      // all waves' fills done; prior reads complete

    if (kk + 1 < 512) {   // prefetch next k-block into the other buffer
      const unsigned nb = (unsigned)((kk + 1) & 1) * TILE_BYTES;
      const unsigned short* a = gA + (size_t)(kk + 1) * 32;
      const unsigned short* b = gB + (size_t)(kk + 1) * 32;
      async_b128(sm + nb + lA,       a);  async_b128(sm + nb + lA + 16u, a + 8);
      async_b128(sm + nb + lB,       b);  async_b128(sm + nb + lB + 16u, b + 8);
    }

    FragAB fa[2], fb[4];
#pragma unroll
    for (int mi = 0; mi < 2; ++mi) {
      unsigned r = (unsigned)(wm * 32 + mi * 16 + (lane & 15));
      unsigned o = bufo + r * ROWSTRIDE + aoff;
      fa[mi].q[0] = smem_q[o >> 4];
      fa[mi].q[1] = smem_q[(o + 32u) >> 4];
    }
#pragma unroll
    for (int ni = 0; ni < 4; ++ni) {
      unsigned r = (unsigned)(wn * 64 + ni * 16 + (lane & 15));
      unsigned o = bufo + HALF_TILE + r * ROWSTRIDE + boff;
      fb[ni].q[0] = smem_q[o >> 4];
      fb[ni].q[1] = smem_q[(o + 32u) >> 4];
    }

#pragma unroll
    for (int mi = 0; mi < 2; ++mi)
#pragma unroll
      for (int ni = 0; ni < 4; ++ni)
        acc[mi][ni] = __builtin_amdgcn_wmma_f32_16x16x32_bf16(
            false, fa[mi].v, false, fb[ni].v, (short)0, acc[mi][ni], false, false);
  }

  // Epilogue: D layout -> row M = r + (lane>=16 ? 8 : 0), col N = lane&15
#pragma unroll
  for (int mi = 0; mi < 2; ++mi) {
#pragma unroll
    for (int ni = 0; ni < 4; ++ni) {
      const int gn = n0 + wn * 64 + ni * 16 + (lane & 15);
      const float bv = bias[gn];
#pragma unroll
      for (int r = 0; r < 8; ++r) {
        const int gm = m0 + wm * 32 + mi * 16 + r + ((lane >> 4) & 1) * 8;
        Cout[(size_t)gm * 4096 + gn] = acc[mi][ni][r] + bv;
      }
    }
  }
}

// ---------------------------------------------------------------------------
// Kernel 4: sequential LIF dynamics + layer 2. One block per batch row.
// v1 kept in registers (16 h-values/thread), spikes gate W2 row sums,
// LDS tree reduction to C=10, v2 + spike count in lanes 0..9.
// ---------------------------------------------------------------------------
__global__ __launch_bounds__(256)
void snn_lif(const float* __restrict__ cur1, const float* __restrict__ W2,
             const float* __restrict__ b2, float* __restrict__ out) {
  const int b = blockIdx.x;
  const int tid = threadIdx.x;
  __shared__ float red[10 * 256];

  float v1[16];
#pragma unroll
  for (int i = 0; i < 16; ++i) v1[i] = 0.f;
  float v2 = 0.f, acc = 0.f;

  for (int t = 0; t < 32; ++t) {
    const float* row = cur1 + (size_t)(b * 32 + t) * 4096;
    float part[10];
#pragma unroll
    for (int c = 0; c < 10; ++c) part[c] = 0.f;

#pragma unroll
    for (int i = 0; i < 16; ++i) {
      const int h = i * 256 + tid;
      const float c1 = row[h];
      v1[i] = 0.5f * (v1[i] + c1);          // v += (cur - v)/TAU, TAU=2
      if (v1[i] >= 1.0f) {                  // spike; hard reset to 0
        v1[i] = 0.f;
        const float* w = W2 + (size_t)h * 10;
#pragma unroll
        for (int c = 0; c < 10; ++c) part[c] += w[c];
      }
    }

#pragma unroll
    for (int c = 0; c < 10; ++c) red[c * 256 + tid] = part[c];
    __syncthreads();
#pragma unroll
    for (int off = 128; off > 0; off >>= 1) {
      if (tid < off) {
#pragma unroll
        for (int c = 0; c < 10; ++c)
          red[c * 256 + tid] += red[c * 256 + tid + off];
      }
      __syncthreads();
    }

    if (tid < 10) {
      const float cur2 = red[tid * 256] + b2[tid];
      v2 = 0.5f * (v2 + cur2);
      if (v2 >= 1.0f) { acc += 1.0f; v2 = 0.f; }
    }
    __syncthreads();
  }

  if (tid < 10) out[b * 10 + tid] = acc * (1.0f / 32.0f);
}

// ---------------------------------------------------------------------------
extern "C" void kernel_launch(void* const* d_in, const int* in_sizes, int n_in,
                              void* d_out, int out_size, void* d_ws, size_t ws_size,
                              hipStream_t stream) {
  const float* x  = (const float*)d_in[0];   // [128,32,16384]
  const float* W1 = (const float*)d_in[1];   // [16384,4096]
  const float* b1 = (const float*)d_in[2];   // [4096]
  const float* W2 = (const float*)d_in[3];   // [4096,10]
  const float* b2 = (const float*)d_in[4];   // [10]
  float* out = (float*)d_out;                // [128,10]

  const size_t ABF_BYTES  = (size_t)4096 * 16384 * 2;  // 128 MB
  const size_t W1T_BYTES  = (size_t)4096 * 16384 * 2;  // 128 MB
  const size_t CUR1_BYTES = (size_t)4096 * 4096 * 4;   //  64 MB
  if (ws_size < ABF_BYTES + W1T_BYTES + CUR1_BYTES) return;

  unsigned short* Abf = (unsigned short*)d_ws;
  unsigned short* W1T = (unsigned short*)((char*)d_ws + ABF_BYTES);
  float* cur1 = (float*)((char*)d_ws + ABF_BYTES + W1T_BYTES);

  const long long nx = (long long)4096 * 16384;
  cvt_f32_bf16_vec4<<<(unsigned)(nx / 1024), 256, 0, stream>>>(x, Abf, nx);
  transpose_cvt<<<dim3(128, 512), dim3(32, 8), 0, stream>>>(W1, W1T);
  snn_gemm_bf16<<<dim3(32, 32), 256, 0, stream>>>(Abf, W1T, b1, cur1);
  snn_lif<<<128, 256, 0, stream>>>(cur1, W2, b2, out);
}
